// MLCAFusion_37125697306965
// MI455X (gfx1250) — matrix operand order repaired and checked by reading, not compiled, not run on hardware
//
#include <hip/hip_runtime.h>
#include <hip/hip_bf16.h>

// ---------------------------------------------------------------------------
// MLCA fusion for MI455X (gfx1250, wave32).  All heavy math on
// v_wmma_f32_16x16x32_f16 (f16 operands, f32 accumulate).
// B=16, Ta=2048, Tv=1024 -> T=1536, D=256, FUSED=512.
//
// All WMMA B operands are either fragment-packed in global memory (conv
// weights) or stored [N,K] so LDS staging is uint4 copy and fragment loads
// vectorize to ds_load_b128.
// ---------------------------------------------------------------------------

typedef __attribute__((ext_vector_type(16))) _Float16 v16h;
typedef __attribute__((ext_vector_type(8)))  float    v8f;

__device__ __forceinline__ float gelu_f(float x) {
    return 0.5f * x * (1.0f + erff(x * 0.70710678118654752440f));
}

// ---- WMMA fragment loaders (contiguous k = 16*(lane/16) + s, A and B alike) --------------------

// A 16x32 f16, src row-major [16, K], row stride ld (halves).
// Per-lane halves are contiguous in groups of 8 -> vectorizes to b128.
__device__ __forceinline__ v16h load_a_frag(const _Float16* src, int ld) {
    int lane  = threadIdx.x & 31;
    int m     = lane & 15;
    int kbase = (lane < 16) ? 0 : 16;
    v16h a;
#pragma unroll
    for (int v = 0; v < 8; ++v) {
        int k = kbase + 2 * v;
        a[2 * v]     = src[m * ld + k];
        a[2 * v + 1] = src[m * ld + k + 1];
    }
    return a;
}

// B 32x16 f16 where source is stored [N, K] (we need B = src^T).
// Per-lane halves contiguous in groups of 16 -> vectorizes to b128.
__device__ __forceinline__ v16h load_bT_frag(const _Float16* src, int ld) {
    int lane  = threadIdx.x & 31;
    int n     = lane & 15;
    int kbase = (lane < 16) ? 0 : 16;
    v16h b;
#pragma unroll
    for (int v = 0; v < 8; ++v) {
        int k = kbase + 2 * v;
        b[2 * v]     = src[n * ld + k];
        b[2 * v + 1] = src[n * ld + k + 1];
    }
    return b;
}

__device__ __forceinline__ v8f wmma_f16(const v16h& a, const v16h& b, const v8f& c) {
    return __builtin_amdgcn_wmma_f32_16x16x32_f16(false, a, false, b, (short)0, c,
                                                  false, false);
}

// ===========================================================================
// 1) Temporal linear interpolation (align_corners=False), f32 + f16 outputs
// ===========================================================================
__global__ void interp_kernel(const float* __restrict__ audio,
                              const float* __restrict__ video,
                              float* __restrict__ aF, float* __restrict__ vF,
                              _Float16* __restrict__ aH, _Float16* __restrict__ vH,
                              int T, int Ta, int Tv, int D) {
    int bt = blockIdx.x;
    int d  = threadIdx.x;
    int b  = bt / T;
    int t  = bt - b * T;
    size_t o = (size_t)bt * D + d;

    { // audio
        float src = (t + 0.5f) * ((float)Ta / (float)T) - 0.5f;
        src = fmaxf(src, 0.0f);
        int i0 = min((int)src, Ta - 1);
        int i1 = min(i0 + 1, Ta - 1);
        float w  = src - (float)i0;
        float x0 = audio[((size_t)b * Ta + i0) * D + d];
        float x1 = audio[((size_t)b * Ta + i1) * D + d];
        float v  = x0 * (1.0f - w) + x1 * w;
        aF[o] = v; aH[o] = (_Float16)v;
    }
    { // video
        float src = (t + 0.5f) * ((float)Tv / (float)T) - 0.5f;
        src = fmaxf(src, 0.0f);
        int i0 = min((int)src, Tv - 1);
        int i1 = min(i0 + 1, Tv - 1);
        float w  = src - (float)i0;
        float x0 = video[((size_t)b * Tv + i0) * D + d];
        float x1 = video[((size_t)b * Tv + i1) * D + d];
        float v  = x0 * (1.0f - w) + x1 * w;
        vF[o] = v; vH[o] = (_Float16)v;
    }
}

// ===========================================================================
// 2) Weight prep.
//    Conv weights [O=128, I=256, Kk] f32 -> WMMA-B-fragment-packed f16:
//      layout [tap][kb(8)][ctile(8)][lane(32)*16 + slot]
//    so a wave loads one fragment as a single contiguous v16h per lane.
// ===========================================================================
__global__ void convw_prep_frag(const float* __restrict__ w,
                                _Float16* __restrict__ out,
                                int I, int Kk) {
    int idx = blockIdx.x * 256 + threadIdx.x;       // over Kk * 8 * 8 * 512
    int total = Kk * 8 * 8 * 512;
    if (idx >= total) return;
    int slot = idx & 15;
    int lane = (idx >> 4) & 31;
    int ct   = (idx >> 9) & 7;                      // cout tile
    int kb   = (idx >> 12) & 7;                     // cin block of 32
    int k    = idx >> 15;                           // tap
    int n     = lane & 15;
    int kbase = (lane < 16) ? 0 : 16;
    int kk    = kbase + slot;                       // cin within block
    int cin   = kb * 32 + kk;
    int cout  = ct * 16 + n;
    out[idx] = (_Float16)w[((size_t)cout * I + cin) * Kk + k];
}

// proj: B^T stored [N,K] == proj_w itself; pure convert.
__global__ void projw_prep(const float* __restrict__ w, _Float16* __restrict__ out) {
    int idx = blockIdx.x * 256 + threadIdx.x;       // over 512*512
    out[idx] = (_Float16)w[idx];
}

// ===========================================================================
// 3) Fused conv1d(k3 & k5) + bias + GELU + L2-normalize.
//    One block = 16 time rows x 256 out channels; 8 waves, 2 WMMA tiles each.
//    Outputs:
//      normH  [B*T, 256] f16  (normalized, row-major; sim operands)
//      multiT [B][256][T] f16 (un-normalized, channel-major; enh B operand)
// ===========================================================================
__global__ void conv_kernel(const _Float16* __restrict__ xh,      // [B*T, 256]
                            const _Float16* __restrict__ w3fp,    // frag-packed
                            const float*    __restrict__ b3,
                            const _Float16* __restrict__ w5fp,    // frag-packed
                            const float*    __restrict__ b5,
                            _Float16* __restrict__ multiT,        // [B][256][T]
                            _Float16* __restrict__ normH,         // [B*T, 256]
                            int T) {
    __shared__ __align__(16) _Float16 ldsX[20 * 256];   // halo rows t0-2 .. t0+17
    __shared__ __align__(16) float    ldsY[16 * 256];   // gelu(conv) tile, f32
    __shared__ float partsq[16 * 16];
    __shared__ float invn[16];

    int tid   = threadIdx.x;
    int grow0 = blockIdx.x * 16;          // global row (b*T + t0)
    int b     = grow0 / T;
    int t0l   = grow0 - b * T;            // local time offset in [0, T)
    size_t bT0 = (size_t)b * T;

    // ---- stage input tile (with zero halo) into LDS: 20 rows x 256 halves
    for (int i = 0; i < 3; ++i) {
        int idx = tid + i * 256;          // 640 uint4 total
        if (idx < 640) {
            int row = idx >> 5;           // 0..19
            int c8  = idx & 31;           // 8-half chunks
            int t   = t0l - 2 + row;
            uint4 val = make_uint4(0u, 0u, 0u, 0u);
            if (t >= 0 && t < T)
                val = *(const uint4*)&xh[(bT0 + t) * 256 + c8 * 8];
            *(uint4*)&ldsX[row * 256 + c8 * 8] = val;
        }
    }
    __syncthreads();

    int wid   = tid >> 5;
    int lane  = tid & 31;
    int nn    = lane & 15;
    int mbase = (lane < 16) ? 0 : 8;

    for (int ti = 0; ti < 2; ++ti) {
        int  tile  = wid * 2 + ti;        // 0..15
        bool is5   = tile >= 8;
        int  ct    = tile & 7;            // cout tile within conv (0..7)
        const _Float16* wfp  = is5 ? w5fp : w3fp;
        const float*    bptr = is5 ? b5  : b3;
        int taps = is5 ? 5 : 3;
        int pad  = is5 ? 2 : 1;

        v8f c = {};
        for (int k = 0; k < taps; ++k) {
            int off = k - pad;            // -2..2
#pragma unroll
            for (int kb = 0; kb < 8; ++kb) {
                v16h a  = load_a_frag(&ldsX[(2 + off) * 256 + kb * 32], 256);
                // fragment-packed weight: one contiguous v16h per lane
                v16h bf = *(const v16h*)&wfp[(((size_t)k * 8 + kb) * 8 + ct) * 512 +
                                             lane * 16];
                c = wmma_f16(a, bf, c);
            }
        }
        int cglob = is5 ? (128 + ct * 16) : (ct * 16);
        float bias = bptr[ct * 16 + nn];
#pragma unroll
        for (int r = 0; r < 8; ++r) {
            float v = gelu_f(c[r] + bias);
            ldsY[(mbase + r) * 256 + cglob + nn] = v;
        }
    }
    __syncthreads();

    // ---- channel-major (transposed) un-normalized output: multiT[b][ch][t]
    {
        int ch = tid;                     // 0..255
        __align__(16) _Float16 tmp[16];
#pragma unroll
        for (int r = 0; r < 16; ++r) tmp[r] = (_Float16)ldsY[r * 256 + ch];
        _Float16* dst = multiT + ((size_t)b * 256 + ch) * T + t0l;
        *(uint4*)&dst[0] = *(uint4*)&tmp[0];
        *(uint4*)&dst[8] = *(uint4*)&tmp[8];
    }

    // ---- per-row L2 norm over 256 channels
    {
        int row = tid >> 4;
        int seg = tid & 15;
        float s = 0.0f;
#pragma unroll
        for (int c = 0; c < 16; ++c) {
            float v = ldsY[row * 256 + seg * 16 + c];
            s += v * v;
        }
        partsq[row * 16 + seg] = s;
    }
    __syncthreads();
    if (tid < 16) {
        float s = 0.0f;
#pragma unroll
        for (int seg = 0; seg < 16; ++seg) s += partsq[tid * 16 + seg];
        float nrm = sqrtf(s);
        invn[tid] = 1.0f / fmaxf(nrm, 1e-12f);
    }
    __syncthreads();
    {
        int row = tid >> 4;
        int seg = tid & 15;
        float iv = invn[row];
        size_t base = (size_t)(grow0 + row) * 256 + seg * 16;
        __align__(16) _Float16 tmp[16];
#pragma unroll
        for (int c = 0; c < 16; ++c)
            tmp[c] = (_Float16)(ldsY[row * 256 + seg * 16 + c] * iv);
        *(uint4*)&normH[base]     = *(uint4*)&tmp[0];
        *(uint4*)&normH[base + 8] = *(uint4*)&tmp[8];
    }
}

// ===========================================================================
// 4) Tiled WMMA GEMM, C[M,N] = A[M,K] * B^T where B is stored [N,K].
//    Block tile 32(M) x 64(N), 8 waves (2x4), K staged in 64-wide LDS chunks.
//    MODE 0: C f32 (sim).  MODE 1: out_h = f16(resid + C) (enh+residual).
//    MODE 2: out_f32 = gelu(C + bias) (projection).
// ===========================================================================
template <int MODE>
__global__ void gemm_kernel(const _Float16* __restrict__ A, int lda,
                            const _Float16* __restrict__ Bm, int ldb,
                            int M, int N, int K,
                            float* __restrict__ out_f32,
                            _Float16* __restrict__ out_h, int ldo, int ncol0,
                            const float* __restrict__ resid, int ldr,
                            const float* __restrict__ bias) {
    __shared__ __align__(16) _Float16 ldsA[32 * 72];
    __shared__ __align__(16) _Float16 ldsB[64 * 72];

    int tid = threadIdx.x;
    int m0  = blockIdx.y * 32;
    int n0  = blockIdx.x * 64;
    int wid = tid >> 5;
    int wr  = wid >> 2;       // 0..1
    int wc  = wid & 3;        // 0..3

    v8f c = {};

    for (int k0 = 0; k0 < K; k0 += 64) {
        // stage A chunk: 32 rows x 64 halves (one uint4 per thread)
        {
            int row = tid >> 3;
            int c8  = tid & 7;
            *(uint4*)&ldsA[row * 72 + c8 * 8] =
                *(const uint4*)&A[(size_t)(m0 + row) * lda + k0 + c8 * 8];
            if (k0 + 64 < K)
                __builtin_prefetch(&A[(size_t)(m0 + row) * lda + k0 + 64 + c8 * 8], 0, 0);
        }
        // stage B chunk (stored [N,K]): 64 rows x 64 halves (two uint4/thread)
#pragma unroll
        for (int i = 0; i < 2; ++i) {
            int idx = tid + i * 256;
            int row = idx >> 3;
            int c8  = idx & 7;
            *(uint4*)&ldsB[row * 72 + c8 * 8] =
                *(const uint4*)&Bm[(size_t)(n0 + row) * ldb + k0 + c8 * 8];
        }
        __syncthreads();

#pragma unroll
        for (int kk = 0; kk < 64; kk += 32) {
            v16h a = load_a_frag(&ldsA[(wr * 16) * 72 + kk], 72);
            v16h b = load_bT_frag(&ldsB[(wc * 16) * 72 + kk], 72);
            c = wmma_f16(a, b, c);
        }
        __syncthreads();
    }

    int lane  = tid & 31;
    int nn    = lane & 15;
    int mbase = (lane < 16) ? 0 : 8;
    int gm0   = m0 + wr * 16 + mbase;
    int gn    = n0 + wc * 16 + nn;
#pragma unroll
    for (int r = 0; r < 8; ++r) {
        int gm = gm0 + r;
        float acc = c[r];
        if (MODE == 0) {
            out_f32[(size_t)gm * ldo + gn] = acc;
        } else if (MODE == 1) {
            out_h[(size_t)gm * ldo + ncol0 + gn] =
                (_Float16)(resid[(size_t)gm * ldr + gn] + acc);
        } else {
            out_f32[(size_t)gm * ldo + gn] = gelu_f(acc + bias[gn]);
        }
    }
}

// ===========================================================================
// 5) Softmax (row and column variants) -> f16 attention weights
// ===========================================================================
__device__ __forceinline__ float block_red_max(float v, float* red) {
    int tid = threadIdx.x;
    red[tid] = v; __syncthreads();
    for (int s = 128; s > 0; s >>= 1) {
        if (tid < s) red[tid] = fmaxf(red[tid], red[tid + s]);
        __syncthreads();
    }
    float r = red[0]; __syncthreads(); return r;
}
__device__ __forceinline__ float block_red_sum(float v, float* red) {
    int tid = threadIdx.x;
    red[tid] = v; __syncthreads();
    for (int s = 128; s > 0; s >>= 1) {
        if (tid < s) red[tid] += red[tid + s];
        __syncthreads();
    }
    float r = red[0]; __syncthreads(); return r;
}

__global__ void softmax_row(const float* __restrict__ sim,
                            _Float16* __restrict__ attn, int T) {
    __shared__ float red[256];
    int t = blockIdx.x, tid = threadIdx.x;
    const float* row = sim + (size_t)t * T;
    float mx = -3.4e38f;
    for (int i = tid; i < T; i += 256) mx = fmaxf(mx, row[i]);
    mx = block_red_max(mx, red);
    float s = 0.0f;
    for (int i = tid; i < T; i += 256) s += expf(row[i] - mx);
    s = block_red_sum(s, red);
    float inv = 1.0f / s;
    for (int i = tid; i < T; i += 256)
        attn[(size_t)t * T + i] = (_Float16)(expf(row[i] - mx) * inv);
}

// column softmax over axis 0; writes transposed so output is row-major attn_v2a
__global__ void softmax_col(const float* __restrict__ sim,
                            _Float16* __restrict__ attn, int T) {
    __shared__ float red[256];
    int j = blockIdx.x, tid = threadIdx.x;
    float mx = -3.4e38f;
    for (int i = tid; i < T; i += 256) mx = fmaxf(mx, sim[(size_t)i * T + j]);
    mx = block_red_max(mx, red);
    float s = 0.0f;
    for (int i = tid; i < T; i += 256) s += expf(sim[(size_t)i * T + j] - mx);
    s = block_red_sum(s, red);
    float inv = 1.0f / s;
    for (int i = tid; i < T; i += 256)
        attn[(size_t)j * T + i] = (_Float16)(expf(sim[(size_t)i * T + j] - mx) * inv);
}

// ===========================================================================
// Host-side orchestration
// ===========================================================================
extern "C" void kernel_launch(void* const* d_in, const int* in_sizes, int n_in,
                              void* d_out, int out_size, void* d_ws, size_t ws_size,
                              hipStream_t stream) {
    (void)in_sizes; (void)n_in; (void)out_size; (void)ws_size;

    const float* audio = (const float*)d_in[0];
    const float* video = (const float*)d_in[1];
    const float* a3w = (const float*)d_in[2];  const float* a3b = (const float*)d_in[3];
    const float* a5w = (const float*)d_in[4];  const float* a5b = (const float*)d_in[5];
    const float* v3w = (const float*)d_in[6];  const float* v3b = (const float*)d_in[7];
    const float* v5w = (const float*)d_in[8];  const float* v5b = (const float*)d_in[9];
    const float* pw  = (const float*)d_in[10]; const float* pb  = (const float*)d_in[11];
    float* out = (float*)d_out;

    const int Bb = 16, T = 1536, Ta = 2048, Tv = 1024, D = 256;
    const size_t BT = (size_t)Bb * T;

    char* p = (char*)d_ws;
    auto alloc = [&](size_t bytes) -> char* {
        char* r = p; p += (bytes + 255) & ~(size_t)255; return r;
    };
    float*    aF      = (float*)   alloc(BT * D * 4);
    float*    vF      = (float*)   alloc(BT * D * 4);
    _Float16* aH      = (_Float16*)alloc(BT * D * 2);
    _Float16* vH      = (_Float16*)alloc(BT * D * 2);
    _Float16* aMultiT = (_Float16*)alloc(BT * D * 2);   // [B][256][T]
    _Float16* vMultiT = (_Float16*)alloc(BT * D * 2);   // [B][256][T]
    _Float16* aNormH  = (_Float16*)alloc(BT * D * 2);   // [B*T, 256]
    _Float16* vNormH  = (_Float16*)alloc(BT * D * 2);   // [B*T, 256]
    _Float16* fusedH  = (_Float16*)alloc(BT * 512 * 2);
    _Float16* w3afp   = (_Float16*)alloc(3 * 256 * 128 * 2);
    _Float16* w5afp   = (_Float16*)alloc(5 * 256 * 128 * 2);
    _Float16* w3vfp   = (_Float16*)alloc(3 * 256 * 128 * 2);
    _Float16* w5vfp   = (_Float16*)alloc(5 * 256 * 128 * 2);
    _Float16* projT   = (_Float16*)alloc(512 * 512 * 2);
    float*    sim     = (float*)   alloc((size_t)T * T * 4);
    _Float16* attnA   = (_Float16*)alloc((size_t)T * T * 2);
    _Float16* attnB   = (_Float16*)alloc((size_t)T * T * 2);

    // 1) interpolation
    interp_kernel<<<dim3(Bb * T), dim3(256), 0, stream>>>(
        audio, video, aF, vF, aH, vH, T, Ta, Tv, D);

    // 2) weight prep (fragment-packed conv weights; f16 convert of proj_w)
    convw_prep_frag<<<dim3(3 * 8 * 8 * 2), dim3(256), 0, stream>>>(a3w, w3afp, 256, 3);
    convw_prep_frag<<<dim3(5 * 8 * 8 * 2), dim3(256), 0, stream>>>(a5w, w5afp, 256, 5);
    convw_prep_frag<<<dim3(3 * 8 * 8 * 2), dim3(256), 0, stream>>>(v3w, w3vfp, 256, 3);
    convw_prep_frag<<<dim3(5 * 8 * 8 * 2), dim3(256), 0, stream>>>(v5w, w5vfp, 256, 5);
    projw_prep<<<dim3(512 * 512 / 256), dim3(256), 0, stream>>>(pw, projT);

    // 3) conv + gelu + normalize (WMMA)
    conv_kernel<<<dim3(Bb * T / 16), dim3(256), 0, stream>>>(
        aH, w3afp, a3b, w5afp, a5b, aMultiT, aNormH, T);
    conv_kernel<<<dim3(Bb * T / 16), dim3(256), 0, stream>>>(
        vH, w3vfp, v3b, w5vfp, v5b, vMultiT, vNormH, T);

    // 4) per-batch cross attention (sim buffer reused; stream order serializes)
    for (int b = 0; b < Bb; ++b) {
        size_t ro = (size_t)b * T * D;       // row offset into [B*T, 256] arrays
        // sim = a_norm @ v_norm^T   [T,T], K=256   (B stored [N,K] = v_norm)
        gemm_kernel<0><<<dim3(T / 64, T / 32), dim3(256), 0, stream>>>(
            aNormH + ro, 256, vNormH + ro, 256, T, T, 256,
            sim, nullptr, T, 0, nullptr, 0, nullptr);

        softmax_row<<<dim3(T), dim3(256), 0, stream>>>(sim, attnA, T);
        softmax_col<<<dim3(T), dim3(256), 0, stream>>>(sim, attnB, T);

        // a_enh = attn_a2v @ v_multi ; fused[:, :256] = a + a_enh
        //   B stored [N=256, K=T] = vMultiT slice for this batch
        gemm_kernel<1><<<dim3(256 / 64, T / 32), dim3(256), 0, stream>>>(
            attnA, T, vMultiT + (size_t)b * 256 * T, T, T, 256, T,
            nullptr, fusedH + (size_t)b * T * 512, 512, 0, aF + ro, 256, nullptr);
        // v_enh = attn_v2a @ a_multi ; fused[:, 256:] = v + v_enh
        gemm_kernel<1><<<dim3(256 / 64, T / 32), dim3(256), 0, stream>>>(
            attnB, T, aMultiT + (size_t)b * 256 * T, T, T, 256, T,
            nullptr, fusedH + (size_t)b * T * 512, 512, 256, vF + ro, 256, nullptr);
    }

    // 5) out = gelu(fused @ proj_w^T + proj_b)   [B*T, 512], K=512
    //    B stored [N,K] = proj_w (f16)
    gemm_kernel<2><<<dim3(512 / 64, (int)(BT / 32)), dim3(256), 0, stream>>>(
        fusedH, 512, projT, 512, (int)BT, 512, 512,
        out, nullptr, 512, 0, nullptr, 0, pb);
}
